// FeCAM_Manager_50371376447968
// MI455X (gfx1250) — matrix operand order repaired and checked by reading, not compiled, not run on hardware
//
#include <hip/hip_runtime.h>

// ---------------- CDNA5 (gfx1250) types ----------------
typedef __bf16 bf16_t;
typedef __attribute__((ext_vector_type(16))) __bf16 v16bf;
typedef __attribute__((ext_vector_type(8)))  float  v8f;
typedef __attribute__((ext_vector_type(4)))  float  v4f;

#define D_DIM   768
#define ROWS    128                 // feature rows per workgroup
#define THREADS 256                 // 8 waves (wave32)
#define WAVES   8
#define STRIDE  776                 // padded bf16 row stride: 1552B = 16B-aligned, banks spread
#define KTILES  (D_DIM / 32)        // 24 WMMA K-steps per e-tile-pair
#define EPAIRS  (D_DIM / 32)        // 24 pairs of 16-wide output-column tiles of Sigma

// two 16B LDS loads -> one 16-element bf16 WMMA operand
__device__ __forceinline__ v16bf ld2(const bf16_t* p0, const bf16_t* p1) {
  union { v16bf v; v4f f[2]; } u;
  u.f[0] = *(const v4f*)p0;
  u.f[1] = *(const v4f*)p1;
  return u.v;
}

__global__ __launch_bounds__(THREADS, 1)
void fecam_mahal_kernel(const float* __restrict__ raw,    // [B, D]
                        const float* __restrict__ means,  // [C, D]
                        const float* __restrict__ covs,   // [C, D, D]
                        float* __restrict__ out,          // [B, C]
                        int B, int C) {
  extern __shared__ __align__(16) char smem[];
  bf16_t* diff  = (bf16_t*)smem;               // ROWS x STRIDE   (198656 B)
  bf16_t* sigma = diff + ROWS * STRIDE;        // 32   x STRIDE   ( 49664 B)
  bf16_t* mnrm  = sigma + 32 * STRIDE;         // D               (  1536 B)
  float*  rrsq  = (float*)(mnrm + D_DIM);      // ROWS            (   512 B)
  __shared__ float s_red[WAVES];

  const int tid  = threadIdx.x;
  const int lane = tid & 31;
  const int wave = tid >> 5;
  const int c    = blockIdx.y;
  const int b0   = blockIdx.x * ROWS;
  const float* mc = means + (size_t)c * D_DIM;
  const float* sg = covs  + (size_t)c * D_DIM * D_DIM;

  // ---- normalize class mean into LDS as bf16 ----
  float ms = 0.f;
  for (int d = tid; d < D_DIM; d += THREADS) { float x = mc[d]; ms += x * x; }
  #pragma unroll
  for (int off = 16; off; off >>= 1) ms += __shfl_xor(ms, off, 32);
  if (lane == 0) s_red[wave] = ms;
  __syncthreads();
  ms = 0.f;
  #pragma unroll
  for (int w = 0; w < WAVES; ++w) ms += s_red[w];
  const float mscale = 1.0f / fmaxf(sqrtf(ms), 1e-12f);
  for (int d = tid; d < D_DIM; d += THREADS) mnrm[d] = (bf16_t)(mc[d] * mscale);

  // ---- per-row Tukey norms: for beta=0.5, t^2 = |x|, so sumsq = sum|x| ----
  for (int rr = wave; rr < ROWS; rr += WAVES) {
    const int b = b0 + rr;
    float s = 0.f;
    if (b < B) {
      const float* row = raw + (size_t)b * D_DIM;
      for (int d = lane; d < D_DIM; d += 32) s += fabsf(row[d]);
    }
    #pragma unroll
    for (int off = 16; off; off >>= 1) s += __shfl_xor(s, off, 32);
    if (lane == 0) rrsq[rr] = 1.0f / fmaxf(sqrtf(s), 1e-12f);
  }
  __syncthreads();

  // ---- stage diff = tukey(raw)/||tukey(raw)|| - m_c  (bf16, padded rows) ----
  for (int i = tid; i < ROWS * D_DIM; i += THREADS) {
    const int r = i / D_DIM, d = i - r * D_DIM;
    const int b = b0 + r;
    float v = 0.f;
    if (b < B) {
      const float x = raw[(size_t)b * D_DIM + d];
      const float t = copysignf(sqrtf(fabsf(x)), x);
      v = t * rrsq[r] - (float)mnrm[d];
    }
    diff[r * STRIDE + d] = (bf16_t)v;
  }

  // ---- fused GEMM + quadratic-form dot over PAIRS of 16-wide e-tiles ----
  // A operand (16x32 bf16): lane&15 = M row; halves hold K 0-15 vs 16-31
  const bf16_t* arow = diff + (wave * 16 + (lane & 15)) * STRIDE;
  const int koff = (lane >> 4) * 16;
  // B operand (32x16 bf16): lane&15 = N col; halves hold K 0-15 vs 16-31
  const bf16_t* brow0 = sigma + (lane & 15) * STRIDE;        // e-tile 0 of pair
  const bf16_t* brow1 = brow0 + 16 * STRIDE;                 // e-tile 1 of pair
  const int bkoff = (lane >> 4) * 16;
  // C/D layout: component r, lane<16 -> M=r ; lane>=16 -> M=8+r ; N = lane&15
  const int mbase = wave * 16 + ((lane >> 4) << 3);
  const int ncol  = lane & 15;

  v8f dotacc = {};
  for (int ep = 0; ep < EPAIRS; ++ep) {
    const int e0 = ep * 32;
    __syncthreads();   // previous sigma block fully consumed
    // stage Sigma[:, e0:e0+32] transposed (f32 -> bf16): sigma[j][d] = Sg[d][e0+j]
    // 32 consecutive lanes read 128B-contiguous rows of Sigma
    for (int i = tid; i < D_DIM * 32; i += THREADS) {
      const int d = i >> 5, j = i & 31;
      sigma[j * STRIDE + d] = (bf16_t)sg[(size_t)d * D_DIM + (e0 + j)];
    }
    __syncthreads();
    // pull next column block toward L2 (global_prefetch_b8)
    if (ep + 1 < EPAIRS) {
      for (int d = tid; d < D_DIM; d += THREADS)
        __builtin_prefetch(sg + (size_t)d * D_DIM + (e0 + 32), 0, 1);
    }

    v8f acc0 = {};
    v8f acc1 = {};
    for (int kt = 0; kt < KTILES; ++kt) {
      const int d = kt * 32;
      v16bf a  = ld2(arow  + d + koff,  arow  + d + koff + 8);   // shared A load
      v16bf b0 = ld2(brow0 + d + bkoff, brow0 + d + bkoff + 8);
      v16bf b1 = ld2(brow1 + d + bkoff, brow1 + d + bkoff + 8);
      acc0 = __builtin_amdgcn_wmma_f32_16x16x32_bf16(
                 false, a, false, b0, (short)0, acc0, false, false);
      acc1 = __builtin_amdgcn_wmma_f32_16x16x32_bf16(
                 false, a, false, b1, (short)0, acc1, false, false);
    }
    // fuse reduction: mahal[b] += Left[b, e] * diff[b, e] for both e-tiles
    #pragma unroll
    for (int r = 0; r < 8; ++r) {
      const bf16_t* drow = diff + (mbase + r) * STRIDE + e0 + ncol;
      dotacc[r] += acc0[r] * (float)drow[0] + acc1[r] * (float)drow[16];
    }
  }

  // reduce 16 e-lanes within each half-wave; lanes 0 / 16 hold row results
  #pragma unroll
  for (int r = 0; r < 8; ++r) {
    #pragma unroll
    for (int off = 8; off; off >>= 1)
      dotacc[r] += __shfl_xor(dotacc[r], off, 32);
  }
  if ((lane & 15) == 0) {
    #pragma unroll
    for (int r = 0; r < 8; ++r) {
      const int b = b0 + mbase + r;
      if (b < B) out[(size_t)b * C + c] = -dotacc[r];
    }
  }
}

extern "C" void kernel_launch(void* const* d_in, const int* in_sizes, int n_in,
                              void* d_out, int out_size, void* d_ws, size_t ws_size,
                              hipStream_t stream) {
  const float* raw   = (const float*)d_in[0];
  const float* means = (const float*)d_in[1];
  const float* covs  = (const float*)d_in[2];
  float* out = (float*)d_out;

  const int D = D_DIM;
  const int B = in_sizes[0] / D;         // 512
  const int C = in_sizes[1] / D;         // 200

  const size_t lds = (size_t)(ROWS * STRIDE + 32 * STRIDE + D_DIM) * sizeof(bf16_t)
                   + (size_t)ROWS * sizeof(float);   // ~250.4 KB < 320 KB/WGP
  (void)hipFuncSetAttribute(reinterpret_cast<const void*>(fecam_mahal_kernel),
                            hipFuncAttributeMaxDynamicSharedMemorySize, (int)lds);

  dim3 grid((B + ROWS - 1) / ROWS, C);   // x fast-varying: same-class blocks share Sigma in L2
  fecam_mahal_kernel<<<grid, dim3(THREADS), lds, stream>>>(raw, means, covs, out, B, C);
}
